// SGMHead_30356828848709
// MI455X (gfx1250) — matrix operand order repaired and checked by reading, not compiled, not run on hardware
//
#include <hip/hip_runtime.h>
#include <hip/hip_bf16.h>
#include <stdint.h>

typedef __attribute__((ext_vector_type(16))) _Float16 v16h;
typedef __attribute__((ext_vector_type(8)))  float    v8f;
typedef unsigned int u32x4 __attribute__((ext_vector_type(4)));
typedef int          i32x4 __attribute__((ext_vector_type(4)));
typedef int          i32x8 __attribute__((ext_vector_type(8)));

#define Bsz 32
#define Lsz 256
#define Ssz 5
#define Nsz 1024
#define Dsz 256
#define DTsz 256
#define Vsz 512
#define Hsz 8
#define DhSz 32

union HF16 { v16h v; _Float16 h[16]; };

// ---------------------------------------------------------------------------
// Tensor Data Mover: 2D tile load (global -> LDS), fp16 elements.
// D# packing per CDNA5 ISA (groups 0/1; 2D so groups
// 2/3 zero). data_size=1 (2 bytes). Tile rows stored compact in LDS.
// ---------------------------------------------------------------------------
__device__ inline void tdm_load_2d_f16(uint32_t ldsAddr, const void* gptr,
                                       int tensorDim0, int tensorDim1,
                                       int tileDim0, int tileDim1, int stride0)
{
  uint64_t ga = (uint64_t)(uintptr_t)gptr;
  u32x4 g0;
  g0[0] = 1u;                                    // count=1, user descriptor
  g0[1] = ldsAddr;                               // lds_addr (bytes)
  g0[2] = (uint32_t)ga;                          // global_addr[31:0]
  g0[3] = (uint32_t)((ga >> 32) & 0x1FFFFFFu) | (2u << 30);  // addr[56:32] | type=2
  i32x8 g1;
  g1[0] = 1 << 16;                               // workgroup_mask=0, data_size=2B
  g1[1] = (tensorDim0 & 0xFFFF) << 16;           // tensor_dim0[15:0]
  g1[2] = ((tensorDim0 >> 16) & 0xFFFF) | ((tensorDim1 & 0xFFFF) << 16);
  g1[3] = ((tensorDim1 >> 16) & 0xFFFF) | ((tileDim0 & 0xFFFF) << 16);
  g1[4] = tileDim1 & 0xFFFF;                     // tile_dim1 (tile_dim2=0)
  g1[5] = stride0;                               // tensor_dim0_stride[31:0]
  g1[6] = 0;
  g1[7] = 0;
  i32x4 gz; gz[0] = 0; gz[1] = 0; gz[2] = 0; gz[3] = 0;
#if defined(__clang_major__) && (__clang_major__ >= 23)
  i32x8 gz8; gz8[0]=0; gz8[1]=0; gz8[2]=0; gz8[3]=0; gz8[4]=0; gz8[5]=0; gz8[6]=0; gz8[7]=0;
  __builtin_amdgcn_tensor_load_to_lds(g0, g1, gz, gz, gz8, 0);
#else
  __builtin_amdgcn_tensor_load_to_lds(g0, g1, gz, gz, 0);
#endif
}

// ---------------------------------------------------------------------------
// fp16 WMMA GEMM with fragment-packed B: C[M,N] = A[M,K] @ B[K,N] (+bias).
// Bpack layout: [kb = K/32][nt = N/16][lane 0..31][16 halves] -- each lane's
// v16h operand is one contiguous aligned 32B load (2x global_load_b128).
// One wave computes a 16x64 strip (A frag reused by 4 back-to-back WMMAs).
// blockDim=128 (4 waves). Strips % 4 == 0.
// ---------------------------------------------------------------------------
__global__ void gemm_h16(const _Float16* __restrict__ A, const _Float16* __restrict__ Bpack,
                         const float* __restrict__ bias, float* __restrict__ C,
                         int M, int N, int K)
{
  int wave = threadIdx.x >> 5;
  int lane = threadIdx.x & 31;
  int lm = lane & 15, lhi = lane >> 4;
  int NT = N >> 4;
  int stripsN = N >> 6;                     // 64-col strips
  int strip = blockIdx.x * 4 + wave;
  int tm = strip / stripsN, sn = strip % stripsN;
  int col0 = sn * 64;
  const _Float16* Arow = A + (size_t)(tm * 16 + lm) * K;
  const v16h* Bp = (const v16h*)Bpack;
  v8f acc0 = {}, acc1 = {}, acc2 = {}, acc3 = {};
#pragma unroll 2
  for (int k0 = 0; k0 < K; k0 += 32) {
    HF16 ua;
#pragma unroll
    for (int v = 0; v < 8; ++v) {
      int kp = k0 + 2 * v + lhi * 16;
      ua.h[2 * v]     = Arow[kp];
      ua.h[2 * v + 1] = Arow[kp + 1];
    }
    int kb = k0 >> 5;
    size_t base = ((size_t)(kb * NT + sn * 4)) * 32 + lane;
    v16h ub0 = Bp[base];
    v16h ub1 = Bp[base + 32];
    v16h ub2 = Bp[base + 64];
    v16h ub3 = Bp[base + 96];
    acc0 = __builtin_amdgcn_wmma_f32_16x16x32_f16(false, ua.v, false, ub0, (short)0, acc0, false, false);
    acc1 = __builtin_amdgcn_wmma_f32_16x16x32_f16(false, ua.v, false, ub1, (short)0, acc1, false, false);
    acc2 = __builtin_amdgcn_wmma_f32_16x16x32_f16(false, ua.v, false, ub2, (short)0, acc2, false, false);
    acc3 = __builtin_amdgcn_wmma_f32_16x16x32_f16(false, ua.v, false, ub3, (short)0, acc3, false, false);
  }
  float b0 = bias ? bias[col0 + lm]      : 0.0f;
  float b1 = bias ? bias[col0 + 16 + lm] : 0.0f;
  float b2 = bias ? bias[col0 + 32 + lm] : 0.0f;
  float b3 = bias ? bias[col0 + 48 + lm] : 0.0f;
#pragma unroll
  for (int j = 0; j < 8; ++j) {
    int m = j + lhi * 8;
    float* Crow = C + (size_t)(tm * 16 + m) * N + col0 + lm;
    Crow[0]  = acc0[j] + b0;
    Crow[16] = acc1[j] + b1;
    Crow[32] = acc2[j] + b2;
    Crow[48] = acc3[j] + b3;
  }
}

// ---------------------------------------------------------------------------
// Conv1d(k=3, pad=1) over S as 3 accumulated GEMMs with on-the-fly embedding
// gather. Conv weights are fragment-packed [t][kb(8)][nt(16)][lane][16]; the
// TDM stages the block's 64-col slab (8 rows x 4KB, stride 16KB) into LDS,
// so each B fragment is one aligned ds_load_b128 pair.
// M = B*L*S = 40960 rows, N = 256, K = 256 per tap.
// ---------------------------------------------------------------------------
__global__ void conv_gemm(const int* __restrict__ ids, const float* __restrict__ emb,
                          const _Float16* __restrict__ Wpack, float* __restrict__ C)
{
  __shared__ alignas(32) _Float16 sB[8 * 2048];   // 32 KB: one tap, 64 cols packed
  int wave = threadIdx.x >> 5, lane = threadIdx.x & 31;
  int lm = lane & 15, lhi = lane >> 4;
  const int tilesN = 16;
  int tile = blockIdx.x * 4 + wave;
  int tm = tile / tilesN, tn = tile % tilesN;
  int tnl = tn & 3;                         // tile within the 64-col group
  int ntBase = tn & ~3;                     // uniform across the 4 waves
  int r = tm * 16 + lm;                     // A-fragment row
  int bl = r / Ssz, s = r % Ssz;
  uint32_t ldsAddr = (uint32_t)(uintptr_t)(void*)&sB[0];
  const v16h* sBp = (const v16h*)sB;
  v8f acc = {};
  for (int t = 0; t < 3; ++t) {
    __syncthreads();                        // previous tap fully consumed
    if (wave == 0) {
      // slab: rows kb=0..7 of 2048 contiguous halves (4 nt groups), stride 8192
      tdm_load_2d_f16(ldsAddr, Wpack + (size_t)t * 65536 + (size_t)ntBase * 512,
                      /*tensorDim0=*/8192, /*tensorDim1=*/8,
                      /*tileDim0=*/2048, /*tileDim1=*/8, /*stride0=*/8192);
      __builtin_amdgcn_s_wait_tensorcnt(0);
    }
    __syncthreads();                        // LDS tile visible to all waves
    int sp = s + t - 1;
    bool valid = (sp >= 0) && (sp < Ssz);
    const float* erow = emb + (size_t)(valid ? ids[bl * Ssz + sp] : 0) * DTsz;
#pragma unroll 2
    for (int k0 = 0; k0 < DTsz; k0 += 32) {
      HF16 ua;
#pragma unroll
      for (int v = 0; v < 8; ++v) {
        int kp = k0 + 2 * v + lhi * 16;
        ua.h[2 * v]     = valid ? (_Float16)erow[kp]     : (_Float16)0.0f;
        ua.h[2 * v + 1] = valid ? (_Float16)erow[kp + 1] : (_Float16)0.0f;
      }
      int kb = k0 >> 5;
      v16h ub = sBp[(kb * 4 + tnl) * 32 + lane];
      acc = __builtin_amdgcn_wmma_f32_16x16x32_f16(false, ua.v, false, ub,
                                                   (short)0, acc, false, false);
    }
  }
#pragma unroll
  for (int j = 0; j < 8; ++j) {
    int m = j + lhi * 8;
    C[(size_t)(tm * 16 + m) * Dsz + tn * 16 + lm] = acc[j];
  }
}

// ---------------------------------------------------------------------------
// Fused cross-attention: one wave per (b, h, 16 query rows). K and V are
// fragment-packed per head, so every B fragment is one aligned v16h load.
// ---------------------------------------------------------------------------
__global__ void attn_kernel(const _Float16* __restrict__ Q, const _Float16* __restrict__ Kpack,
                            const _Float16* __restrict__ Vpack, const float* __restrict__ mask,
                            const float* __restrict__ tau, _Float16* __restrict__ F)
{
  __shared__ float sS[16 * Nsz];      // 64 KB
  int lane = threadIdx.x & 31;
  int lm = lane & 15, lhi = lane >> 4;
  int blk = blockIdx.x;
  int lt = blk & 15;                  // L/16 = 16
  int h  = (blk >> 4) & 7;
  int b  = blk >> 7;
  float tc = fminf(fmaxf(tau[0], 0.25f), 4.0f);
  float scale = 1.0f / (5.65685424949238f * tc);   // 1/(sqrt(32)*tau_c)

  HF16 qa;                            // A frag: Q tile [16 x 32]
  {
    const _Float16* Ap = Q + (size_t)(b * Lsz + lt * 16 + lm) * Dsz + h * DhSz;
#pragma unroll
    for (int v = 0; v < 8; ++v) {
      int kp = 2 * v + lhi * 16;
      qa.h[2 * v] = Ap[kp]; qa.h[2 * v + 1] = Ap[kp + 1];
    }
  }
  const v16h* Kp = (const v16h*)Kpack + ((size_t)(b * Hsz + h) * 64) * 32 + lane;
  const v16h* Vp = (const v16h*)Vpack + ((size_t)(b * Hsz + h) * 64) * 32 + lane;
#pragma unroll 2
  for (int nt = 0; nt < Nsz / 16; ++nt) {
    v16h kb = Kp[(size_t)nt * 32];    // B frag: K^T [32 x 16]
    v8f sc = {};
    sc = __builtin_amdgcn_wmma_f32_16x16x32_f16(false, qa.v, false, kb,
                                                (short)0, sc, false, false);
    int n = nt * 16 + lm;
#pragma unroll
    for (int j = 0; j < 8; ++j) {
      int m = j + lhi * 8;
      float msk = mask[(size_t)(b * Lsz + lt * 16 + m) * Nsz + n];
      sS[m * Nsz + n] = sc[j] * scale * msk;
    }
  }
  __syncthreads();
  if (lane < 16) {                    // one lane per query row
    float* row = sS + lane * Nsz;
    float mx = -3.4e38f;
    for (int n = 0; n < Nsz; ++n) mx = fmaxf(mx, row[n]);
    float sum = 0.f;
    for (int n = 0; n < Nsz; ++n) { float e = __expf(row[n] - mx); row[n] = e; sum += e; }
    float inv = 1.0f / sum;
    for (int n = 0; n < Nsz; ++n) row[n] *= inv;
  }
  __syncthreads();
  v8f o0 = {}, o1 = {};
#pragma unroll 2
  for (int kt = 0; kt < Nsz / 32; ++kt) {
    HF16 pa;                          // A frag from probabilities in LDS
    const float* Pp = sS + lm * Nsz + kt * 32;
#pragma unroll
    for (int v = 0; v < 8; ++v) {
      int kp = 2 * v + lhi * 16;
      pa.h[2 * v] = (_Float16)Pp[kp]; pa.h[2 * v + 1] = (_Float16)Pp[kp + 1];
    }
    v16h vb0 = Vp[(size_t)(kt * 2) * 32];
    v16h vb1 = Vp[(size_t)(kt * 2 + 1) * 32];
    o0 = __builtin_amdgcn_wmma_f32_16x16x32_f16(false, pa.v, false, vb0,
                                                (short)0, o0, false, false);
    o1 = __builtin_amdgcn_wmma_f32_16x16x32_f16(false, pa.v, false, vb1,
                                                (short)0, o1, false, false);
  }
#pragma unroll
  for (int j = 0; j < 8; ++j) {
    int m = j + lhi * 8;
    size_t base = (size_t)(b * Lsz + lt * 16 + m) * Dsz + h * DhSz;
    F[base + lm]      = (_Float16)o0[j];
    F[base + 16 + lm] = (_Float16)o1[j];
  }
}

// ---------------------------------------------------------------------------
// LayerNorm over last dim (256), fp32 in -> fp16 out. One block per row.
// ---------------------------------------------------------------------------
__global__ void ln_h16(const float* __restrict__ X, const float* __restrict__ g,
                       const float* __restrict__ bt, _Float16* __restrict__ Y)
{
  __shared__ float red[256];
  int row = blockIdx.x, tid = threadIdx.x;
  float v = X[(size_t)row * 256 + tid];
  red[tid] = v; __syncthreads();
  for (int s = 128; s > 0; s >>= 1) { if (tid < s) red[tid] += red[tid + s]; __syncthreads(); }
  float mean = red[0] * (1.0f / 256.0f); __syncthreads();
  float d = v - mean;
  red[tid] = d * d; __syncthreads();
  for (int s = 128; s > 0; s >>= 1) { if (tid < s) red[tid] += red[tid + s]; __syncthreads(); }
  float var = red[0] * (1.0f / 256.0f);
  Y[(size_t)row * 256 + tid] = (_Float16)(d * rsqrtf(var + 1e-5f) * g[tid] + bt[tid]);
}

// Faithful torch flat-reshape mean: q[bl,j] = (1/5) sum_i conv[(bl*5 + f%5), f/5], f=i*256+j
__global__ void scramble_mean(const float* __restrict__ Cc, const float* __restrict__ dir,
                              _Float16* __restrict__ out)
{
  int bl = blockIdx.x, j = threadIdx.x;
  float sum = 0.f;
#pragma unroll
  for (int i = 0; i < 5; ++i) {
    int f = i * 256 + j;
    int dt = f / 5, s = f - dt * 5;
    sum += Cc[((size_t)bl * 5 + s) * 256 + dt];
  }
  out[(size_t)bl * 256 + j] = (_Float16)(sum * 0.2f + dir[j]);
}

// Weight -> fragment-packed B. Source W row-major [N,K] (so B[k][n] = W[n][k]).
// out[((kb*NT + nt)*32 + lane)*16 + e], e=2j+par -> k = kb*32 + lhi*16 + 2j+par.
__global__ void pack_b_trans(const float* __restrict__ W, _Float16* __restrict__ out,
                             int N, int K)
{
  int idx = blockIdx.x * 256 + threadIdx.x;
  if (idx >= N * K) return;
  int e = idx & 15;
  int lane = (idx >> 4) & 31;
  int rest = idx >> 9;
  int NT = N >> 4;
  int nt = rest % NT, kb = rest / NT;
  int j = e >> 1, par = e & 1;
  int lm = lane & 15, lhi = lane >> 4;
  int k = kb * 32 + lhi * 16 + 2 * j + par;
  int n = nt * 16 + lm;
  out[idx] = (_Float16)W[(size_t)n * K + k];
}

__global__ void cvt_h16(const float* __restrict__ in, _Float16* __restrict__ out, int n)
{
  int idx = blockIdx.x * 256 + threadIdx.x;
  if (idx < n) out[idx] = (_Float16)in[idx];
}

// conv_w (o,i,t) -> fragment-packed per tap: out[t][kb(8)][nt(16)][lane][16]
__global__ void pack_b_conv(const float* __restrict__ w, _Float16* __restrict__ out)
{
  int idx = blockIdx.x * 256 + threadIdx.x;        // 196608 total
  int e = idx & 15;
  int lane = (idx >> 4) & 31;
  int rest = idx >> 9;                              // (t*8 + kb)*16 + nt
  int nt = rest & 15;
  int kb = (rest >> 4) & 7;
  int t = rest >> 7;
  int j = e >> 1, par = e & 1;
  int lm = lane & 15, lhi = lane >> 4;
  int k = kb * 32 + lhi * 16 + 2 * j + par;         // input channel i
  int n = nt * 16 + lm;                             // output channel o
  out[idx] = (_Float16)w[(size_t)(n * 256 + k) * 3 + t];
}

// hK [B*N, D] -> score-B fragments: out[b][h][nt(64)][lane][16]
// frag element: k(dh) = lhi*16 + 2j+par, n(token) = nt*16+lm
__global__ void pack_kh(const _Float16* __restrict__ Kh, _Float16* __restrict__ out)
{
  int idx = blockIdx.x * 256 + threadIdx.x;        // 8388608 total
  int e = idx & 15;
  int lane = (idx >> 4) & 31;
  int nt = (idx >> 9) & 63;
  int h = (idx >> 15) & 7;
  int b = idx >> 18;
  int j = e >> 1, par = e & 1;
  int lm = lane & 15, lhi = lane >> 4;
  int k = lhi * 16 + 2 * j + par;
  int n = nt * 16 + lm;
  out[idx] = Kh[(size_t)(b * Nsz + n) * Dsz + h * DhSz + k];
}

// hV [B*N, D] -> PV-B fragments: out[b][h][kb(32)][nt(2)][lane][16]
// frag element: k(token) = kb*32 + lhi*16 + 2j+par, col = nt*16+lm
__global__ void pack_vh(const _Float16* __restrict__ Vh, _Float16* __restrict__ out)
{
  int idx = blockIdx.x * 256 + threadIdx.x;        // 8388608 total
  int e = idx & 15;
  int lane = (idx >> 4) & 31;
  int nt = (idx >> 9) & 1;
  int kb = (idx >> 10) & 31;
  int h = (idx >> 15) & 7;
  int b = idx >> 18;
  int j = e >> 1, par = e & 1;
  int lm = lane & 15, lhi = lane >> 4;
  int k = kb * 32 + lhi * 16 + 2 * j + par;
  int col = nt * 16 + lm;
  out[idx] = Vh[(size_t)(b * Nsz + k) * Dsz + h * DhSz + col];
}

// Per-(b,l) softmax(logits/2) agreement term (one block each).
__global__ void agree_k(const float* __restrict__ Ll, const float* __restrict__ Lr,
                        float* __restrict__ pairOut)
{
  __shared__ float red[256];
  int b = blockIdx.x / 255, l = blockIdx.x % 255, tid = threadIdx.x;
  const float* rl = Ll + (size_t)(b * Lsz + l) * Vsz;
  const float* rr = Lr + (size_t)(b * Lsz + l + 1) * Vsz;
  float a0 = rl[tid] * 0.5f, a1 = rl[tid + 256] * 0.5f;
  float c0 = rr[tid] * 0.5f, c1 = rr[tid + 256] * 0.5f;
  red[tid] = fmaxf(a0, a1); __syncthreads();
  for (int s = 128; s > 0; s >>= 1) { if (tid < s) red[tid] = fmaxf(red[tid], red[tid + s]); __syncthreads(); }
  float mL = red[0]; __syncthreads();
  red[tid] = fmaxf(c0, c1); __syncthreads();
  for (int s = 128; s > 0; s >>= 1) { if (tid < s) red[tid] = fmaxf(red[tid], red[tid + s]); __syncthreads(); }
  float mR = red[0]; __syncthreads();
  float e0 = __expf(a0 - mL), e1 = __expf(a1 - mL);
  float f0 = __expf(c0 - mR), f1 = __expf(c1 - mR);
  red[tid] = e0 + e1; __syncthreads();
  for (int s = 128; s > 0; s >>= 1) { if (tid < s) red[tid] += red[tid + s]; __syncthreads(); }
  float sL = red[0]; __syncthreads();
  red[tid] = f0 + f1; __syncthreads();
  for (int s = 128; s > 0; s >>= 1) { if (tid < s) red[tid] += red[tid + s]; __syncthreads(); }
  float sR = red[0]; __syncthreads();
  float iL = 1.0f / sL, iR = 1.0f / sR;
  float d0 = e0 * iL - f0 * iR, d1 = e1 * iL - f1 * iR;
  red[tid] = d0 * d0 + d1 * d1; __syncthreads();
  for (int s = 128; s > 0; s >>= 1) { if (tid < s) red[tid] += red[tid + s]; __syncthreads(); }
  if (tid == 0) pairOut[blockIdx.x] = red[0];
}

// Deterministic single-block final reduction: loss = 0.1*sum(agree*mask)/max(sum(mask),1)
__global__ void loss_final(const float* __restrict__ pairOut, const float* __restrict__ mask,
                           float* __restrict__ out)
{
  __shared__ float red[256];
  int tid = threadIdx.x;
  float num = 0.f;
  for (int i = tid; i < 32 * 255; i += 256) {
    int b = i / 255, l = i % 255;
    num += pairOut[i] * mask[b * Lsz + l];
  }
  red[tid] = num; __syncthreads();
  for (int s = 128; s > 0; s >>= 1) { if (tid < s) red[tid] += red[tid + s]; __syncthreads(); }
  num = red[0]; __syncthreads();
  float den = 0.f;
  for (int i = tid; i < 32 * 255; i += 256) {
    int b = i / 255, l = i % 255;
    den += mask[b * Lsz + l];
  }
  red[tid] = den; __syncthreads();
  for (int s = 128; s > 0; s >>= 1) { if (tid < s) red[tid] += red[tid + s]; __syncthreads(); }
  if (tid == 0) out[0] = 0.1f * num / fmaxf(red[0], 1.0f);
}

extern "C" void kernel_launch(void* const* d_in, const int* in_sizes, int n_in,
                              void* d_out, int out_size, void* d_ws, size_t ws_size,
                              hipStream_t stream)
{
  (void)in_sizes; (void)n_in; (void)out_size; (void)ws_size;
  const float* vis   = (const float*)d_in[0];
  const int*   idsL  = (const int*)d_in[1];
  const int*   idsR  = (const int*)d_in[2];
  /* d_in[3] tgt_ids unused by the reference math */
  const float* tmask = (const float*)d_in[4];
  const float* vmask = (const float*)d_in[5];
  const float* embw  = (const float*)d_in[6];
  const float* dirL  = (const float*)d_in[7];
  const float* dirR  = (const float*)d_in[8];
  const float* txtW  = (const float*)d_in[9];
  const float* txtB  = (const float*)d_in[10];
  const float* qG    = (const float*)d_in[11];
  const float* qB    = (const float*)d_in[12];
  const float* kW    = (const float*)d_in[13];
  const float* kB    = (const float*)d_in[14];
  const float* vW    = (const float*)d_in[15];
  const float* vB    = (const float*)d_in[16];
  const float* kvG   = (const float*)d_in[17];
  const float* kvB   = (const float*)d_in[18];
  const float* tau   = (const float*)d_in[19];
  const float* clsB  = (const float*)d_in[20];
  const float* cwL   = (const float*)d_in[21];
  const float* cwR   = (const float*)d_in[22];

  float* out = (float*)d_out;
  float* logitsL = out + 1;
  float* logitsR = out + 1 + (size_t)Bsz * Lsz * Vsz;

  char* p = (char*)d_ws;
  auto alloc = [&](size_t bytes) { void* r = (void*)p; p += (bytes + 255) & ~(size_t)255; return r; };
  _Float16* hVis   = (_Float16*)alloc((size_t)Bsz * Nsz * Dsz * 2);
  _Float16* hK     = (_Float16*)alloc((size_t)Bsz * Nsz * Dsz * 2);
  _Float16* hV     = (_Float16*)alloc((size_t)Bsz * Nsz * Dsz * 2);
  _Float16* hKpack = (_Float16*)alloc((size_t)Bsz * Nsz * Dsz * 2);
  _Float16* hVpack = (_Float16*)alloc((size_t)Bsz * Nsz * Dsz * 2);
  _Float16* hEmbT  = (_Float16*)alloc((size_t)Vsz * DTsz * 2);
  _Float16* hTxtW  = (_Float16*)alloc((size_t)Dsz * DTsz * 2);
  _Float16* hKW    = (_Float16*)alloc((size_t)Dsz * Dsz * 2);
  _Float16* hVW    = (_Float16*)alloc((size_t)Dsz * Dsz * 2);
  _Float16* hCW    = (_Float16*)alloc((size_t)3 * Dsz * DTsz * 2);
  _Float16* hQpre  = (_Float16*)alloc((size_t)Bsz * Lsz * DTsz * 2);
  _Float16* hQ     = (_Float16*)alloc((size_t)Bsz * Lsz * Dsz * 2);
  _Float16* hF     = (_Float16*)alloc((size_t)Bsz * Lsz * Dsz * 2);
  float* fBig      = (float*)alloc((size_t)Bsz * Lsz * Ssz * Dsz * 4);   // 40960x256
  float* fPair     = (float*)alloc((size_t)Bsz * (Lsz - 1) * 4);

  // fp16 staging: weights go straight to fragment-packed B layout
  pack_b_trans<<<(Dsz * DTsz + 255) / 256, 256, 0, stream>>>(txtW, hTxtW, Dsz, DTsz);
  pack_b_trans<<<(Dsz * Dsz + 255) / 256, 256, 0, stream>>>(kW, hKW, Dsz, Dsz);
  pack_b_trans<<<(Dsz * Dsz + 255) / 256, 256, 0, stream>>>(vW, hVW, Dsz, Dsz);
  pack_b_trans<<<(Vsz * DTsz + 255) / 256, 256, 0, stream>>>(embw, hEmbT, Vsz, DTsz);
  cvt_h16<<<(Bsz * Nsz * Dsz + 255) / 256, 256, 0, stream>>>(vis, hVis, Bsz * Nsz * Dsz);

  // Shared K/V projections + LN + head-major fragment packing (once for both sides)
  {
    int M = Bsz * Nsz;
    int strips = (M / 16) * (Dsz / 64);
    gemm_h16<<<strips / 4, 128, 0, stream>>>(hVis, hKW, kB, fBig, M, Dsz, Dsz);
    ln_h16<<<M, 256, 0, stream>>>(fBig, kvG, kvB, hK);
    gemm_h16<<<strips / 4, 128, 0, stream>>>(hVis, hVW, vB, fBig, M, Dsz, Dsz);
    ln_h16<<<M, 256, 0, stream>>>(fBig, kvG, kvB, hV);
    pack_kh<<<(Bsz * Nsz * Dsz) / 256, 256, 0, stream>>>(hK, hKpack);
    pack_vh<<<(Bsz * Nsz * Dsz) / 256, 256, 0, stream>>>(hV, hVpack);
  }

  for (int side = 0; side < 2; ++side) {
    const int* ids   = side ? idsR : idsL;
    const float* dir = side ? dirR : dirL;
    const float* cw  = side ? cwR : cwL;
    float* logits    = side ? logitsR : logitsL;

    pack_b_conv<<<(3 * Dsz * DTsz) / 256, 256, 0, stream>>>(cw, hCW);
    {
      int tiles = (Bsz * Lsz * Ssz / 16) * (Dsz / 16);   // 40960 tiles
      conv_gemm<<<tiles / 4, 128, 0, stream>>>(ids, embw, hCW, fBig);
    }
    scramble_mean<<<Bsz * Lsz, 256, 0, stream>>>(fBig, dir, hQpre);
    {
      int M = Bsz * Lsz;
      int strips = (M / 16) * (Dsz / 64);
      gemm_h16<<<strips / 4, 128, 0, stream>>>(hQpre, hTxtW, txtB, fBig, M, Dsz, DTsz);
      ln_h16<<<M, 256, 0, stream>>>(fBig, qG, qB, hQ);
    }
    attn_kernel<<<Bsz * Hsz * (Lsz / 16), 32, 0, stream>>>(hQ, hKpack, hVpack, vmask, tau, hF);
    {
      int M = Bsz * Lsz;
      int strips = (M / 16) * (Vsz / 64);
      gemm_h16<<<strips / 4, 128, 0, stream>>>(hF, hEmbT, clsB, logits, M, Vsz, Dsz);
    }
  }

  agree_k<<<Bsz * (Lsz - 1), 256, 0, stream>>>(logitsL, logitsR, fPair);
  loss_final<<<1, 256, 0, stream>>>(fPair, tmask, out);
}
